// VectorQuantizer_31129922961670
// MI455X (gfx1250) — matrix operand order repaired and checked by reading, not compiled, not run on hardware
//
#include <hip/hip_runtime.h>

typedef __attribute__((ext_vector_type(16))) _Float16 v16h;
typedef __attribute__((ext_vector_type(8)))  _Float16 v8h;
typedef __attribute__((ext_vector_type(8)))  float    v8f;

#define NROWS   16384      // 16 * 32 * 32
#define KCODES  8192
#define DFULL   256
#define TN      64         // rows per block tile
#define TKG     64         // codes per group tile (4 col tiles)
#define KCHUNK  2048       // codes per block (grid.y = 4)
#define LDSS    264        // padded LDS row stride in halves (528B = 132 DW -> 4-bank lane stride)
#define SCALE_E 4096.0f
#define QCOUNT  4194304    // 16*256*32*32
#define IOFF    4194304
#define LOSSOFF 4210688

#if __has_builtin(__builtin_amdgcn_global_load_async_to_lds_b128)
#define HAS_ASYNC_LDS 1
typedef int i4v __attribute__((vector_size(4 * sizeof(int))));
typedef __attribute__((address_space(1))) i4v* g_i4p;   // global
typedef __attribute__((address_space(3))) i4v* l_i4p;   // LDS
#else
#define HAS_ASYNC_LDS 0
#endif

// copy one 16B unit global->LDS (async if available)
__device__ __forceinline__ void copy16_g2l(_Float16* ldst, const _Float16* gsrc) {
#if HAS_ASYNC_LDS
    __builtin_amdgcn_global_load_async_to_lds_b128((g_i4p)gsrc, (l_i4p)ldst, 0, 0);
#else
    *(uint4*)ldst = *(const uint4*)gsrc;
#endif
}

__device__ __forceinline__ void wait_g2l() {
#if HAS_ASYNC_LDS
#if __has_builtin(__builtin_amdgcn_s_wait_asynccnt)
    __builtin_amdgcn_s_wait_asynccnt(0);
#else
    asm volatile("s_wait_asynccnt 0x0" ::: "memory");
#endif
#endif
}

__device__ __forceinline__ v16h cat8(v8h a, v8h b) {
    return __builtin_shufflevector(a, b, 0,1,2,3,4,5,6,7,8,9,10,11,12,13,14,15);
}

// B fragment (32x16 f16): lane = code column, runs [d0 + hi*8, +8) and [d0 + 16 + hi*8, +8)
__device__ __forceinline__ v16h loadB(const _Float16* brow, int d0, int hi) {
    return cat8(*(const v8h*)(brow + d0 + hi * 8),
                *(const v8h*)(brow + d0 + 16 + hi * 8));
}

__device__ __forceinline__ unsigned int fkey(float f) {
    unsigned int b = __float_as_uint(f);
    return b ^ ((b & 0x80000000u) ? 0xFFFFFFFFu : 0x80000000u);
}

__device__ __forceinline__ unsigned long long shfl_xor_u64(unsigned long long v, int mask) {
    unsigned int lo = (unsigned int)v;
    unsigned int hi = (unsigned int)(v >> 32);
    lo = (unsigned int)__shfl_xor((int)lo, mask, 32);
    hi = (unsigned int)__shfl_xor((int)hi, mask, 32);
    return ((unsigned long long)hi << 32) | lo;
}

// ---- kernel 1: z [16,256,1024] fp32 -> zf16 [16384, 256] f16 (LDS transpose) ----
__global__ __launch_bounds__(256) void convert_z(const float* __restrict__ z,
                                                 _Float16* __restrict__ zf) {
    __shared__ float tile[32][33];
    const int b  = blockIdx.z;
    const int d0 = blockIdx.y * 32;
    const int h0 = blockIdx.x * 32;
    const int tx = threadIdx.x & 31;
    const int ty = threadIdx.x >> 5;   // 0..7
    #pragma unroll
    for (int i = 0; i < 32; i += 8) {
        int d = d0 + ty + i;
        tile[ty + i][tx] = z[(((size_t)b * 256 + d) << 10) + h0 + tx];
    }
    __syncthreads();
    #pragma unroll
    for (int i = 0; i < 32; i += 8) {
        int hw = h0 + ty + i;
        zf[(((size_t)b * 1024 + hw) << 8) + d0 + tx] = (_Float16)tile[tx][ty + i];
    }
}

// ---- kernel 2: emb -> scaled f16 + fp32 row norms ----
__global__ __launch_bounds__(256) void convert_e(const float* __restrict__ emb,
                                                 _Float16* __restrict__ ef,
                                                 float* __restrict__ enorm) {
    const int k = blockIdx.x;
    const int d = threadIdx.x;
    float v = emb[(size_t)k * DFULL + d];
    ef[(size_t)k * DFULL + d] = (_Float16)(v * SCALE_E);
    __shared__ float red[256];
    red[d] = v * v;
    __syncthreads();
    for (int s = 128; s > 0; s >>= 1) {
        if (d < s) red[d] += red[d + s];
        __syncthreads();
    }
    if (d == 0) enorm[k] = red[0];
}

// ---- kernel 3: init keys + zero loss ----
__global__ __launch_bounds__(256) void vq_init(unsigned long long* __restrict__ keys,
                                               float* __restrict__ out) {
    int i = blockIdx.x * 256 + threadIdx.x;
    if (i < NROWS) keys[i] = ~0ULL;
    if (i == 0) out[LOSSOFF] = 0.0f;
}

// ---- kernel 4: WMMA score GEMM + fused argmin ----
// Per block: 64 rows x 2048 codes. A strip held in registers (8 x v16h per wave);
// one 33.8KB LDS buffer reused: Z tile once, then E tile per 64-code group.
// B fragments software-pipelined one step ahead so ds_loads overlap the WMMAs.
__global__ __launch_bounds__(256) void vq_main(const _Float16* __restrict__ zf,
                                               const _Float16* __restrict__ ef,
                                               const float* __restrict__ enorm,
                                               unsigned long long* __restrict__ keys) {
    __shared__ __align__(16) _Float16 LDSBUF[TKG * LDSS];   // 33,792 B (TN == TKG)

    const int t      = threadIdx.x;
    const int wave   = t >> 5;
    const int lane   = t & 31;
    const int lrow   = lane & 15;
    const int hi     = lane >> 4;          // half-wave select
    const int rstrip = wave & 3;           // 16-row strip of the 64-row tile
    const int cpair  = wave >> 2;          // which pair of 16-code tiles (0 or 1)
    const int n_base = blockIdx.x * TN;
    const int k_chunk = blockIdx.y * KCHUNK;

    // ---- stage Z tile (64 x 256 halves) into LDS, coalesced 16B units ----
    for (int u = t; u < TN * (DFULL / 8); u += 256) {
        int row = u >> 5;
        int col = (u & 31) * 8;
        copy16_g2l(&LDSBUF[row * LDSS + col],
                   &zf[((size_t)(n_base + row) << 8) + col]);
    }
    wait_g2l();
    __syncthreads();

    // ---- pull this wave's A strip into registers: 8 x v16h = 64 VGPRs ----
    v16h Areg[8];
    {
        const _Float16* arow = &LDSBUF[(rstrip * 16 + lrow) * LDSS];
        #pragma unroll
        for (int dd = 0; dd < 8; ++dd) {
            const _Float16* p = arow + dd * 32;
            Areg[dd] = cat8(*(const v8h*)(p + hi * 8),
                            *(const v8h*)(p + 16 + hi * 8));
        }
    }
    __syncthreads();   // everyone has A; LDS buffer is now free for E tiles

    float bestS[8];
    int   bestC[8];
    #pragma unroll
    for (int j = 0; j < 8; ++j) { bestS[j] = 3.4e38f; bestC[j] = 0; }

    const _Float16* b0row = &LDSBUF[((cpair * 2 + 0) * 16 + lrow) * LDSS];
    const _Float16* b1row = &LDSBUF[((cpair * 2 + 1) * 16 + lrow) * LDSS];
    const float c2 = 2.0f / SCALE_E;

    for (int g = 0; g < KCHUNK / TKG; ++g) {
        const int k_base = k_chunk + g * TKG;
        // stage 64-code E tile (async to LDS)
        for (int u = t; u < TKG * (DFULL / 8); u += 256) {
            int row = u >> 5;
            int col = (u & 31) * 8;
            copy16_g2l(&LDSBUF[row * LDSS + col],
                       &ef[((size_t)(k_base + row) << 8) + col]);
        }
        wait_g2l();
        __syncthreads();

        v8f acc0 = {};
        v8f acc1 = {};
        // software pipeline: prefetch next-step B fragments into distinct regs
        v16h B0 = loadB(b0row, 0, hi);
        v16h B1 = loadB(b1row, 0, hi);
        #pragma unroll
        for (int dd = 0; dd < 8; ++dd) {
            v16h C0, C1;
            if (dd < 7) {
                C0 = loadB(b0row, (dd + 1) * 32, hi);
                C1 = loadB(b1row, (dd + 1) * 32, hi);
            }
            acc0 = __builtin_amdgcn_wmma_f32_16x16x32_f16(
                       false, Areg[dd], false, B0, (short)0, acc0, false, false);
            acc1 = __builtin_amdgcn_wmma_f32_16x16x32_f16(
                       false, Areg[dd], false, B1, (short)0, acc1, false, false);
            if (dd < 7) { B0 = C0; B1 = C1; }
        }

        const int col0 = k_base + (cpair * 2) * 16 + lrow;   // global code index (tile 0)
        const int col1 = col0 + 16;                          // tile 1
        const float en0 = enorm[col0];
        const float en1 = enorm[col1];
        #pragma unroll
        for (int j = 0; j < 8; ++j) {
            float s0 = en0 - c2 * acc0[j];   // ||e||^2 - 2 z.e  (row-const dropped)
            if (s0 < bestS[j]) { bestS[j] = s0; bestC[j] = col0; }
            float s1 = en1 - c2 * acc1[j];
            if (s1 < bestS[j]) { bestS[j] = s1; bestC[j] = col1; }
        }
        __syncthreads();   // protect LDS buffer before next group's stores
    }

    // reduce best over the 16 lanes holding the same output row, then global atomic min
    #pragma unroll
    for (int j = 0; j < 8; ++j) {
        unsigned long long key =
            ((unsigned long long)fkey(bestS[j]) << 32) | (unsigned int)bestC[j];
        #pragma unroll
        for (int m = 8; m >= 1; m >>= 1) {
            unsigned long long o = shfl_xor_u64(key, m);
            if (o < key) key = o;
        }
        if (lrow == 0) {
            int row = n_base + rstrip * 16 + j + hi * 8;
            atomicMin(&keys[row], key);
        }
    }
}

// ---- kernel 5: gather quantized, write indices, commit loss ----
__global__ __launch_bounds__(256) void vq_finalize(const float* __restrict__ z,
                                                   const float* __restrict__ emb,
                                                   const unsigned long long* __restrict__ keys,
                                                   float* __restrict__ out) {
    const int o  = blockIdx.x * 256 + threadIdx.x;   // grid exactly covers QCOUNT
    const int hw = o & 1023;
    const int bd = o >> 10;
    const int d  = bd & 255;
    const int b  = bd >> 8;
    const int n  = (b << 10) | hw;
    const unsigned idx = (unsigned)keys[n];          // low 32 bits = code index
    const float q = emb[(size_t)idx * DFULL + d];
    out[o] = q;                                      // straight-through forward == quantized
    if (d == 0) out[IOFF + n] = (float)idx;
    float diff = z[o] - q;
    __shared__ float red[256];
    red[threadIdx.x] = diff * diff;
    __syncthreads();
    for (int s = 128; s > 0; s >>= 1) {
        if (threadIdx.x < s) red[threadIdx.x] += red[threadIdx.x + s];
        __syncthreads();
    }
    if (threadIdx.x == 0)
        atomicAdd(&out[LOSSOFF], red[0] * (1.0f / (float)QCOUNT));
}

extern "C" void kernel_launch(void* const* d_in, const int* in_sizes, int n_in,
                              void* d_out, int out_size, void* d_ws, size_t ws_size,
                              hipStream_t stream) {
    const float* z   = (const float*)d_in[0];   // [16,256,32,32]
    const float* emb = (const float*)d_in[1];   // [8192,256]
    float* out = (float*)d_out;

    char* ws = (char*)d_ws;
    _Float16* zf = (_Float16*)ws;                                   // 8 MB
    _Float16* ef = (_Float16*)(ws + 8388608);                       // 4 MB
    float* enorm = (float*)(ws + 8388608 + 4194304);                // 32 KB
    unsigned long long* keys =
        (unsigned long long*)(ws + 8388608 + 4194304 + 32768);      // 128 KB

    convert_z<<<dim3(32, 8, 16), 256, 0, stream>>>(z, zf);
    convert_e<<<KCODES, 256, 0, stream>>>(emb, ef, enorm);
    vq_init<<<NROWS / 256, 256, 0, stream>>>(keys, out);
    vq_main<<<dim3(NROWS / TN, KCODES / KCHUNK), 256, 0, stream>>>(zf, ef, enorm, keys);
    vq_finalize<<<QCOUNT / 256, 256, 0, stream>>>(z, emb, keys, out);
}
